// MoE_25005299597538
// MI455X (gfx1250) — compile-verified, matrix operand order checked by reading, not run on hardware
//
#include <hip/hip_runtime.h>
#include <hip/hip_bf16.h>

// ---------------------------------------------------------------------------
// MoE: out[n,o] = sum_e w[n,e] * ( relu(relu(x@W1_e+b1)@W2_e+b2) @ W3_e + b3 )
// bf16 WMMA (v_wmma_f32_16x16x32_bf16), f32 accumulate, per-expert streaming.
// GEMM mainloop: double-buffered async global->LDS staging (ASYNCcnt) when
// the gfx1250 async builtins are available; direct-global fallback otherwise.
// ---------------------------------------------------------------------------

typedef __attribute__((ext_vector_type(16))) __bf16 bf16x16;
typedef __attribute__((ext_vector_type(8)))  __bf16 bf16x8;
typedef __attribute__((ext_vector_type(8)))  float  f32x8;
typedef int v4i __attribute__((__vector_size__(4 * sizeof(int))));

#define DEVINL __device__ __forceinline__

#if defined(__gfx1250__) && __has_builtin(__builtin_amdgcn_global_load_async_to_lds_b128)
#define USE_ASYNC_LDS 1
#else
#define USE_ASYNC_LDS 0
#endif

#if USE_ASYNC_LDS
DEVINL void async_copy16(const __bf16* g, __bf16* l) {
  __builtin_amdgcn_global_load_async_to_lds_b128(
      (v4i*)const_cast<__bf16*>(g), (v4i*)l, 0, 0);
}

DEVINL void wait_async0() {
#if __has_builtin(__builtin_amdgcn_s_wait_asynccnt)
  __builtin_amdgcn_s_wait_asynccnt(0);
#else
  asm volatile("s_wait_asynccnt 0" ::: "memory");
#endif
}
#endif

// Padded LDS row stride for 32-wide K tiles: 40 elements = 80B, so the
// 16-row x 16B fragment read pattern spreads across all 64 banks.
#define LDS_STRIDE 40

// Load a 16x32 bf16 fragment (A layout; B identical when weights are stored
// transposed [Nout][K]) per ISA 16-bit A 16x32 VGPR layout:
//   lane L -> row = row0 + L%16
//   elements 0..7  <- K = 8*(L/16) .. +7 ; elements 8..15 <- +16
DEVINL bf16x16 load_frag_g(const __bf16* __restrict__ base, int ld,
                           int row0, int k0, int lane) {
  const int r = row0 + (lane & 15);
  const int k = k0 + ((lane >> 4) << 3);
  const __bf16* p = base + (size_t)r * (size_t)ld + k;
  union { bf16x16 v; bf16x8 h[2]; } u;
  u.h[0] = *reinterpret_cast<const bf16x8*>(p);
  u.h[1] = *reinterpret_cast<const bf16x8*>(p + 16);
  return u.v;
}

DEVINL bf16x16 load_frag_lds(const __bf16* base, int row0, int lane) {
  const int r = row0 + (lane & 15);
  const int k = (lane >> 4) << 3;
  const __bf16* p = base + r * LDS_STRIDE + k;
  union { bf16x16 v; bf16x8 h[2]; } u;
  u.h[0] = *reinterpret_cast<const bf16x8*>(p);
  u.h[1] = *reinterpret_cast<const bf16x8*>(p + 16);
  return u.v;
}

// ---------------------------------------------------------------------------
// GEMM: C[M,N] = A[M,K](bf16) @ Bt[N,K](bf16, pre-transposed) + bias[N]
// COMBINE=false: C -> relu -> bf16 store.
// COMBINE=true : Cf[m,n] (+)= wprob[m*8+expert] * (acc + bias)   (f32 out)
// Block: 256 thr = 8 waves; block tile 128(M) x 256(N); wave tile 64x64.
// ---------------------------------------------------------------------------
template <bool COMBINE>
__global__ __launch_bounds__(256) void moe_gemm_kernel(
    const __bf16* __restrict__ A, const __bf16* __restrict__ Bt,
    const float* __restrict__ bias,
    __bf16* __restrict__ Cb, float* __restrict__ Cf,
    const float* __restrict__ wprob,
    int M, int N, int K, int expert, int init) {
  const int lane = threadIdx.x & 31;
  const int wave = threadIdx.x >> 5;
  const int bm0 = blockIdx.x * 128;
  const int bn0 = blockIdx.y * 256;
  const int wmL = (wave & 1) * 64;   // wave's M offset within block tile
  const int wnL = (wave >> 1) * 64;  // wave's N offset within block tile
  const int wm0 = bm0 + wmL;
  const int wn0 = bn0 + wnL;

  f32x8 acc[4][4] = {};

#if USE_ASYNC_LDS
  __shared__ __bf16 As[2][128 * LDS_STRIDE];
  __shared__ __bf16 Bs[2][256 * LDS_STRIDE];
  const int tid = threadIdx.x;

  // Stage a 128x32 A tile and 256x32 B tile (16B chunks) into one LDS buffer.
  auto stage = [&](int k0, __bf16* as, __bf16* bs) {
#pragma unroll
    for (int t = 0; t < 2; ++t) {            // A: 512 chunks, 2/thread
      const int s = tid + t * 256;
      const int row = s >> 2, c = (s & 3) << 3;
      async_copy16(A + (size_t)(bm0 + row) * K + k0 + c, as + row * LDS_STRIDE + c);
    }
#pragma unroll
    for (int t = 0; t < 4; ++t) {            // B: 1024 chunks, 4/thread
      const int s = tid + t * 256;
      const int row = s >> 2, c = (s & 3) << 3;
      async_copy16(Bt + (size_t)(bn0 + row) * K + k0 + c, bs + row * LDS_STRIDE + c);
    }
  };

  const int nk = K >> 5;
  stage(0, As[0], Bs[0]);
  for (int it = 0; it < nk; ++it) {
    const int cur = it & 1;
    wait_async0();        // my async writes into buffer `cur` have landed
    __syncthreads();      // everyone's writes landed; prior compute finished
    if (it + 1 < nk)      // overlap next tile fill with this tile's WMMAs
      stage((it + 1) << 5, As[cur ^ 1], Bs[cur ^ 1]);

    bf16x16 a[4], b[4];
#pragma unroll
    for (int i = 0; i < 4; ++i) a[i] = load_frag_lds(As[cur], wmL + 16 * i, lane);
#pragma unroll
    for (int j = 0; j < 4; ++j) b[j] = load_frag_lds(Bs[cur], wnL + 16 * j, lane);
#pragma unroll
    for (int i = 0; i < 4; ++i)
#pragma unroll
      for (int j = 0; j < 4; ++j)
        acc[i][j] = __builtin_amdgcn_wmma_f32_16x16x32_bf16(
            false, a[i], false, b[j], (short)0, acc[i][j], false, false);
  }
#else
  for (int k0 = 0; k0 < K; k0 += 32) {
    bf16x16 a[4], b[4];
#pragma unroll
    for (int i = 0; i < 4; ++i) a[i] = load_frag_g(A, K, wm0 + 16 * i, k0, lane);
#pragma unroll
    for (int j = 0; j < 4; ++j) b[j] = load_frag_g(Bt, K, wn0 + 16 * j, k0, lane);
#pragma unroll
    for (int i = 0; i < 4; ++i)
#pragma unroll
      for (int j = 0; j < 4; ++j)
        acc[i][j] = __builtin_amdgcn_wmma_f32_16x16x32_bf16(
            false, a[i], false, b[j], (short)0, acc[i][j], false, false);
  }
#endif

  // C/D layout: VGPR r, lane L -> M = r + 8*(L/16), N = L%16.
  const int nlo = lane & 15;
  const int mhi = (lane >> 4) << 3;

  if (!COMBINE) {
#pragma unroll
    for (int j = 0; j < 4; ++j) {
      const int n = wn0 + 16 * j + nlo;
      const float bs = bias[n];
#pragma unroll
      for (int i = 0; i < 4; ++i)
#pragma unroll
        for (int r = 0; r < 8; ++r) {
          const int m = wm0 + 16 * i + mhi + r;
          float v = acc[i][j][r] + bs;
          v = v > 0.f ? v : 0.f;
          Cb[(size_t)m * (size_t)N + n] = (__bf16)v;
        }
    }
  } else {
    float wv[4][8];
#pragma unroll
    for (int i = 0; i < 4; ++i)
#pragma unroll
      for (int r = 0; r < 8; ++r)
        wv[i][r] = wprob[(size_t)(wm0 + 16 * i + mhi + r) * 8 + expert];
#pragma unroll
    for (int j = 0; j < 4; ++j) {
      const int n = wn0 + 16 * j + nlo;
      const float bs = bias[n];
#pragma unroll
      for (int i = 0; i < 4; ++i)
#pragma unroll
        for (int r = 0; r < 8; ++r) {
          const int m = wm0 + 16 * i + mhi + r;
          const float v = (acc[i][j][r] + bs) * wv[i][r];
          const size_t off = (size_t)m * (size_t)N + n;
          if (init) Cf[off] = v;
          else      Cf[off] += v;
        }
    }
  }
}

// ---------------------------------------------------------------------------
// Gate: scores = (x@gW + gb)/e ; softmax ; keep top-5 of 8 ; renormalize.
// One wave per token.
// ---------------------------------------------------------------------------
__global__ __launch_bounds__(256) void gate_kernel(
    const float* __restrict__ x, const float* __restrict__ gW,
    const float* __restrict__ gb, float* __restrict__ wprob, int Ntok, int D) {
  const int lane = threadIdx.x & 31;
  const int token = blockIdx.x * (blockDim.x >> 5) + (threadIdx.x >> 5);
  if (token >= Ntok) return;

  float s[8];
#pragma unroll
  for (int e = 0; e < 8; ++e) s[e] = 0.f;

  const float* xr = x + (size_t)token * D;
  for (int d = lane; d < D; d += 32) {
    const float xv = xr[d];
    const float* g = gW + (size_t)d * 8;
#pragma unroll
    for (int e = 0; e < 8; ++e) s[e] += xv * g[e];
  }
#pragma unroll
  for (int o = 16; o > 0; o >>= 1)
#pragma unroll
    for (int e = 0; e < 8; ++e) s[e] += __shfl_xor(s[e], o);

  const float inv_temp = 0.36787944117144233f;  // 1/e
  float p[8], mx = -3.4e38f;
#pragma unroll
  for (int e = 0; e < 8; ++e) {
    p[e] = (s[e] + gb[e]) * inv_temp;
    mx = p[e] > mx ? p[e] : mx;
  }
  float sum = 0.f;
#pragma unroll
  for (int e = 0; e < 8; ++e) { p[e] = __expf(p[e] - mx); sum += p[e]; }
  const float isum = 1.f / sum;
#pragma unroll
  for (int e = 0; e < 8; ++e) p[e] *= isum;

  // Drop the 3 smallest (== keep top-5). On ties prefer dropping later idx.
  bool keep[8];
#pragma unroll
  for (int e = 0; e < 8; ++e) keep[e] = true;
  for (int t = 0; t < 3; ++t) {
    int mi = 0; float mv = 3.4e38f;
    for (int e = 0; e < 8; ++e)
      if (keep[e] && p[e] <= mv) { mv = p[e]; mi = e; }
    keep[mi] = false;
  }
  float ks = 0.f;
#pragma unroll
  for (int e = 0; e < 8; ++e) ks += keep[e] ? p[e] : 0.f;
  const float inv = 1.f / (ks + 1e-8f);
  if (lane < 8)
    wprob[(size_t)token * 8 + lane] = keep[lane] ? p[lane] * inv : 0.f;
}

// f32 -> bf16 elementwise
__global__ __launch_bounds__(256) void cast_kernel(
    const float* __restrict__ in, __bf16* __restrict__ out, size_t n) {
  size_t i = (size_t)blockIdx.x * blockDim.x + threadIdx.x;
  if (i < n) out[i] = (__bf16)in[i];
}

// in[IR][IC] f32 -> out[IC][IR] bf16  (coalesced writes)
__global__ __launch_bounds__(256) void transpose_cast_kernel(
    const float* __restrict__ in, __bf16* __restrict__ out, int IR, int IC) {
  size_t o = (size_t)blockIdx.x * blockDim.x + threadIdx.x;
  if (o >= (size_t)IR * IC) return;
  const int r = (int)(o % IR);
  const int c = (int)(o / IR);
  out[o] = (__bf16)in[(size_t)r * IC + c];
}

// ---------------------------------------------------------------------------
extern "C" void kernel_launch(void* const* d_in, const int* in_sizes, int n_in,
                              void* d_out, int out_size, void* d_ws,
                              size_t ws_size, hipStream_t stream) {
  const float* x  = (const float*)d_in[0];
  const float* gW = (const float*)d_in[1];
  const float* gb = (const float*)d_in[2];
  const float* W1 = (const float*)d_in[3];
  const float* b1 = (const float*)d_in[4];
  const float* W2 = (const float*)d_in[5];
  const float* b2 = (const float*)d_in[6];
  const float* W3 = (const float*)d_in[7];
  const float* b3 = (const float*)d_in[8];
  float* out = (float*)d_out;

  const int N = 8192, D = 1024, H = 2048, O = 1024, E = 8;

  char* ws = (char*)d_ws;
  size_t off = 0;
  auto alloc = [&](size_t bytes) -> void* {
    void* p = ws + off;
    off += bytes;
    off = (off + 255) & ~(size_t)255;
    return p;
  };
  __bf16* xb    = (__bf16*)alloc((size_t)N * D * 2);
  float*  wprob = (float*) alloc((size_t)N * E * 4);
  __bf16* w1t   = (__bf16*)alloc((size_t)H * D * 2);
  __bf16* w2t   = (__bf16*)alloc((size_t)H * H * 2);
  __bf16* w3t   = (__bf16*)alloc((size_t)O * H * 2);
  __bf16* h1    = (__bf16*)alloc((size_t)N * H * 2);
  __bf16* h2    = (__bf16*)alloc((size_t)N * H * 2);

  cast_kernel<<<(N * D + 255) / 256, 256, 0, stream>>>(x, xb, (size_t)N * D);
  gate_kernel<<<N / 8, 256, 0, stream>>>(x, gW, gb, wprob, N, D);

  for (int e = 0; e < E; ++e) {
    transpose_cast_kernel<<<(D * H + 255) / 256, 256, 0, stream>>>(
        W1 + (size_t)e * D * H, w1t, D, H);
    transpose_cast_kernel<<<(H * H + 255) / 256, 256, 0, stream>>>(
        W2 + (size_t)e * H * H, w2t, H, H);
    transpose_cast_kernel<<<(H * O + 255) / 256, 256, 0, stream>>>(
        W3 + (size_t)e * H * O, w3t, H, O);

    dim3 g1(N / 128, H / 256);
    moe_gemm_kernel<false><<<g1, 256, 0, stream>>>(
        xb, w1t, b1 + (size_t)e * H, h1, nullptr, nullptr, N, H, D, e, 0);
    dim3 g2(N / 128, H / 256);
    moe_gemm_kernel<false><<<g2, 256, 0, stream>>>(
        h1, w2t, b2 + (size_t)e * H, h2, nullptr, nullptr, N, H, H, e, 0);
    dim3 g3(N / 128, O / 256);
    moe_gemm_kernel<true><<<g3, 256, 0, stream>>>(
        h2, w3t, b3 + (size_t)e * O, nullptr, out, wprob, N, O, H, e,
        e == 0 ? 1 : 0);
  }
}
